// HDMultiheadAttention_18270790877548
// MI455X (gfx1250) — matrix operand order repaired and checked by reading, not compiled, not run on hardware
//
#include <hip/hip_runtime.h>

// ---------------------------------------------------------------------------
// HD (binary) multi-head attention, fused, WMMA f16 path for gfx1250.
//   S=2048, B=2, E=1024, H=16, D=64, heads G = B*H = 32
//
//   attn[n,m] = ( sum_d query_n[d]*query_m[d]*s_q[d]*s_k[d] ) > 0   in {0,1}
//   out[n,d]  = sign( sum_m attn[n,m]*query_m[d] ) * s_v[d]*s_o[d]
//
// Sign algebra: s_q*s_k folded into register-resident Q fragments, s_v*s_o
// into the epilogue -> streamed LDS tiles are raw f16 query data.
//
// Layout trick: the score matmul is computed TRANSPOSED (S^T = K x Q^T) so
// its C/D register layout coincides exactly with the A-fragment layout the
// P x V matmul needs: binarized scores feed the second WMMA with zero
// cross-lane data movement (no LDS restage).
// ---------------------------------------------------------------------------

typedef __attribute__((ext_vector_type(16))) _Float16 v16h;
typedef __attribute__((ext_vector_type(8)))  _Float16 v8h;
typedef __attribute__((ext_vector_type(8)))  float    v8f;
typedef __attribute__((ext_vector_type(4)))  float    v4f;

#define S_LEN      2048
#define NHEAD      16
#define HDIM       64
#define ROW_STRIDE 2048      // B*E floats per sequence position
#define MTILE      128       // query rows per workgroup (8 waves x 16)
#define NTILE      128       // kv rows per LDS tile
#define KT_PAD     72        // K-tile row stride (halves), padded vs 64
#define VT_PAD     136       // transposed V-tile row stride (halves), padded vs 128

static __device__ __forceinline__ float fsign(float x) {
    return (x > 0.f) ? 1.f : ((x < 0.f) ? -1.f : 0.f);
}

static __device__ __forceinline__ v16h cat16(v8h a, v8h b) {
    v16h r;
#pragma unroll
    for (int i = 0; i < 8; ++i) { r[i] = a[i]; r[i + 8] = b[i]; }
    return r;
}

static __device__ __forceinline__ v8f wmma_f16(v16h a, v16h b, v8f c) {
    // D = A(16x32 f16) x B(32x16 f16) + C(16x16 f32)
    return __builtin_amdgcn_wmma_f32_16x16x32_f16(
        /*neg_a=*/false, a, /*neg_b=*/false, b,
        /*c_mod=*/(short)0, c, /*reuse_a=*/false, /*reuse_b=*/false);
}

__global__ __launch_bounds__(256)
void hdmha_fused_kernel(const float* __restrict__ query,
                        const float* __restrict__ w_q,
                        const float* __restrict__ w_k,
                        const float* __restrict__ w_v,
                        const float* __restrict__ w_o,
                        float* __restrict__ out)
{
    __shared__ __align__(32) _Float16 ktile[NTILE * KT_PAD];     // 18 KB, raw q rows
    __shared__ __align__(32) _Float16 vtileT[HDIM * VT_PAD];     // 17 KB, transposed

    const int tid   = threadIdx.x;
    const int wave  = tid >> 5;
    const int lane  = tid & 31;
    const int lhalf = lane >> 4;       // 0 | 1  (lane group)
    const int lmod  = lane & 15;

    const int g = blockIdx.y;          // global head 0..31  (g = b*16 + h)
    const int h = g & (NHEAD - 1);     // head index into weight vectors
    const int rowbase = blockIdx.x * MTILE + wave * 16;

    const float* __restrict__ qhead = query + (size_t)g * HDIM;  // +row*ROW_STRIDE+d
    const float* __restrict__ wq = w_q + h * HDIM;
    const float* __restrict__ wk = w_k + h * HDIM;
    const float* __restrict__ wv = w_v + h * HDIM;
    const float* __restrict__ wo = w_o + h * HDIM;

    // ---- register-resident Q as B-fragments (for S^T = K x Q^T) ----------
    // B 32x16 f16 layout: lane L holds N = L%16 (its own query row);
// slot i<8 -> K(d) = 8*(L/16) + i, slot i>=8 -> 16 + 8*(L/16) + (i-8).  bq1: +32.
    // sign(w_q)*sign(w_k) folded in per d.
    v16h bq0, bq1;
    {
        const float* qp = qhead + (size_t)(rowbase + lmod) * ROW_STRIDE;
        const int db = lhalf * 8;
#pragma unroll
        for (int i = 0; i < 16; ++i) {
            const int d0 = db + i + (i & 8);
            const int d1 = 32 + db + i + (i & 8);
            bq0[i] = (_Float16)((wq[d0] * wk[d0] > 0.f) ? qp[d0] : -qp[d0]);
            bq1[i] = (_Float16)((wq[d1] * wk[d1] > 0.f) ? qp[d1] : -qp[d1]);
        }
    }

    v8f oacc[4];
    const v8f vzero = {0.f, 0.f, 0.f, 0.f, 0.f, 0.f, 0.f, 0.f};
#pragma unroll
    for (int i = 0; i < 4; ++i) oacc[i] = vzero;

    // tile-load geometry fixed per thread (256 threads * 8 iters * float4)
    const int c4    = tid & 15;    // float4 column within the 64-wide head
    const int rbase = tid >> 4;    // base row (stride 16 per iteration)

    for (int kvbase = 0; kvbase < S_LEN; kvbase += NTILE) {
        __syncthreads();   // protect LDS tiles from previous iteration's readers

        // ---- cooperative raw-query tile load (feeds both layouts) --------
#pragma unroll
        for (int it = 0; it < 8; ++it) {
            const int row = rbase + it * 16;                       // 0..127
            const float* src = qhead + (size_t)(kvbase + row) * ROW_STRIDE + c4 * 4;
            const v4f ld = *(const v4f*)src;
            _Float16 hv[4];
#pragma unroll
            for (int j = 0; j < 4; ++j) hv[j] = (_Float16)ld[j];
            // row-major copy (4 contiguous halves -> single b64 store)
#pragma unroll
            for (int j = 0; j < 4; ++j) ktile[row * KT_PAD + c4 * 4 + j] = hv[j];
            // transposed copy (padded stride keeps stores conflict-free)
#pragma unroll
            for (int j = 0; j < 4; ++j) vtileT[(c4 * 4 + j) * VT_PAD + row] = hv[j];
        }
        // prefetch next tile (emits global_prefetch_b8; no counter impact)
        if (kvbase + NTILE < S_LEN) {
            __builtin_prefetch(qhead + (size_t)(kvbase + NTILE + rbase) * ROW_STRIDE + c4 * 4, 0, 0);
        }
        __syncthreads();

        // ---- per-wave compute over four 32-kv chunks ----------------------
#pragma unroll
        for (int kc = 0; kc < 4; ++kc) {
            const int kvc = kc * 32;

            // S^T tiles: sT0 = kv rows [kvc,kvc+16), sT1 = [kvc+16,kvc+32)
            // A 16x32 f16 layout for K rows: lane L: M(kv) = L%16;
            // slot i<8 -> d = 8*(L/16)+i, slot i>=8 -> d = 16+8*(L/16)+(i-8).
            const int ab = lhalf * 8;
            const int r0 = (kvc + lmod) * KT_PAD;
            const int r1 = (kvc + 16 + lmod) * KT_PAD;
            v16h ak;
            ak = cat16(*(const v8h*)&ktile[r0 + ab],
                       *(const v8h*)&ktile[r0 + 16 + ab]);
            v8f sT0 = wmma_f16(ak, bq0, vzero);
            ak = cat16(*(const v8h*)&ktile[r0 + 32 + ab],
                       *(const v8h*)&ktile[r0 + 48 + ab]);
            sT0 = wmma_f16(ak, bq1, sT0);
            ak = cat16(*(const v8h*)&ktile[r1 + ab],
                       *(const v8h*)&ktile[r1 + 16 + ab]);
            v8f sT1 = wmma_f16(ak, bq0, vzero);
            ak = cat16(*(const v8h*)&ktile[r1 + 32 + ab],
                       *(const v8h*)&ktile[r1 + 48 + ab]);
            sT1 = wmma_f16(ak, bq1, sT1);

            // binarize: S^T C/D layout (lane = n, reg r = kv r+8*lhalf) is
            // EXACTLY the A-fragment layout of P[n][kv] -> no data movement.
            v16h ap;
#pragma unroll
            for (int r = 0; r < 8; ++r) {
                ap[r]     = (sT0[r] > 0.f) ? (_Float16)1.f : (_Float16)0.f;
                ap[r + 8] = (sT1[r] > 0.f) ? (_Float16)1.f : (_Float16)0.f;
            }

            // O[16 x 64] += P[16 x 32] x V[32 x 64]  (four 16-wide d chunks)
            // B 32x16 layout from vtileT: lane: d-col = L%16; slot i<8: kv = 8*lhalf+i, i>=8: kv = 16+8*lhalf+(i-8)
#pragma unroll
            for (int dc = 0; dc < 4; ++dc) {
                const int drow = dc * 16 + lmod;
                const int kvo  = kvc + lhalf * 8;
                v16h bv = cat16(*(const v8h*)&vtileT[drow * VT_PAD + kvo],
                                *(const v8h*)&vtileT[drow * VT_PAD + kvo + 16]);
                oacc[dc] = wmma_f16(ap, bv, oacc[dc]);
            }
        }
    }

    // ---- epilogue: out = sign(acc) * sign(w_v) * sign(w_o) ----------------
#pragma unroll
    for (int dc = 0; dc < 4; ++dc) {
        const int d = dc * 16 + lmod;
        const float svo = (wv[d] * wo[d] > 0.f) ? 1.f : -1.f;   // s_v*s_o (never 0)
#pragma unroll
        for (int r = 0; r < 8; ++r) {
            const int row = rowbase + r + lhalf * 8;
            out[(size_t)row * ROW_STRIDE + g * HDIM + d] = fsign(oacc[dc][r]) * svo;
        }
    }
}

extern "C" void kernel_launch(void* const* d_in, const int* in_sizes, int n_in,
                              void* d_out, int out_size, void* d_ws, size_t ws_size,
                              hipStream_t stream) {
    const float* query = (const float*)d_in[0];
    const float* w_q   = (const float*)d_in[1];
    const float* w_k   = (const float*)d_in[2];
    const float* w_v   = (const float*)d_in[3];
    const float* w_o   = (const float*)d_in[4];
    float* out = (float*)d_out;

    dim3 grid(S_LEN / MTILE, 32);   // 16 row-blocks x 32 heads
    hdmha_fused_kernel<<<grid, 256, 0, stream>>>(query, w_q, w_k, w_v, w_o, out);
}
